// Attention_22119081574796
// MI455X (gfx1250) — matrix operand order repaired and checked by reading, not compiled, not run on hardware
//
#include <hip/hip_runtime.h>
#include <hip/hip_bf16.h>

typedef __attribute__((ext_vector_type(16))) _Float16 v16h;
typedef __attribute__((ext_vector_type(8)))  float    v8f;
typedef __attribute__((ext_vector_type(2)))  _Float16 h2;
typedef int vsi4 __attribute__((vector_size(16)));   // matches builtin param

// ---------------------------------------------------------------------------
// CDNA5 async global->LDS copy (GLOBAL_LOAD_ASYNC_TO_LDS_B128, ASYNCcnt).
// ROCm clang declares it as (v4i AS1*, v4i AS3*, imm offset, imm cpol)
// (diagnosed from the round-2 compile error).  Guarded by __has_builtin with a
// VGPR-staged fallback so the file compiles on either toolchain.
#if __has_builtin(__builtin_amdgcn_global_load_async_to_lds_b128)
#define HAVE_ASYNC_LDS 1
#endif

__device__ __forceinline__ void async_copy16(const void* g, void* l) {
#ifdef HAVE_ASYNC_LDS
  __builtin_amdgcn_global_load_async_to_lds_b128(
      (__attribute__((address_space(1))) vsi4*)const_cast<void*>(g),
      (__attribute__((address_space(3))) vsi4*)l, 0, 0);
#else
  *(uint4*)l = *(const uint4*)g;
#endif
}

template <int N>
__device__ __forceinline__ void wait_async() {
#ifdef HAVE_ASYNC_LDS
#if __has_builtin(__builtin_amdgcn_s_wait_asynccnt)
  __builtin_amdgcn_s_wait_asynccnt(N);
#else
  asm volatile("s_wait_asynccnt %0" ::"i"(N) : "memory");
#endif
#endif
}

// ---------------------------------------------------------------------------
// WMMA fragment loaders for V_WMMA_F32_16X16X32_F16 (wave32).
// A (16x32, MxK): lane<16 -> M=lane (K offset 0), lane>=16 -> M=lane-16
// (K offset +8); VGPR r pair base = (r<4 ? 2r : 16+2(r-4)).  Pairs contiguous
// when the tile is row-major [M][K] -> 2x ds_load_b128 per lane.
__device__ __forceinline__ v16h load_frag_a(const _Float16* __restrict__ T,
                                            int rowBase, int lane) {
  v16h f;
  const int m    = rowBase + (lane & 15);
  const int koff = (lane & 16) ? 8 : 0;
  const _Float16* row = T + m * 32;
#pragma unroll
  for (int r = 0; r < 8; ++r) {
    const int kb = ((r < 4) ? 2 * r : 16 + 2 * (r - 4)) + koff;
    h2 p = *(const h2*)(row + kb);
    f[2 * r] = p[0];
    f[2 * r + 1] = p[1];
  }
  return f;
}

// B (32x16, KxN): same K order as A: lane<16 K offset 0, lane>=16 K offset
// +8; VGPR r pair base = (r<4 ? 2r : 16+2(r-4)).  Pairs contiguous when the tile is stored [N][K]
// (W's natural row-major order, since B = W^T).
__device__ __forceinline__ v16h load_frag_b(const _Float16* __restrict__ T,
                                            int nBase, int lane) {
  v16h f;
  const int n    = nBase + (lane & 15);
  const int koff = (lane & 16) ? 8 : 0;
  const _Float16* row = T + n * 32;
#pragma unroll
  for (int r = 0; r < 8; ++r) {
    h2 p = *(const h2*)(row + koff + ((r < 4) ? 2 * r : 16 + 2 * (r - 4)));
    f[2 * r] = p[0];
    f[2 * r + 1] = p[1];
  }
  return f;
}

#define WMMA_F16(A, B, C)                                                     \
  __builtin_amdgcn_wmma_f32_16x16x32_f16(false, (A), false, (B), (short)0,    \
                                         (C), false, false)

// ---------------------------------------------------------------------------
// fp32 -> fp16 conversion
__global__ void f32_to_f16(const float* __restrict__ in,
                           _Float16* __restrict__ out, size_t n) {
  size_t i = (size_t)blockIdx.x * blockDim.x + threadIdx.x;
  size_t stride = (size_t)gridDim.x * blockDim.x;
  for (; i < n; i += stride) out[i] = (_Float16)in[i];
}

// ---------------------------------------------------------------------------
// Tiled GEMM:  out = A @ W^T,  A: rows x 1024 (f16), W: 1024x1024 (f16).
// BM=128, BN=64, BK=32, 256 threads = 8 waves; wave w owns rows 16w..16w+15
// and all 4 column tiles (one head when MODE is 1/2).  LDS tiles are
// double-buffered and filled with async global->LDS copies: tile t+1 is in
// flight (ASYNCcnt<=3 per wave) while tile t feeds the WMMAs.
// MODE 0: plain, f32 out[outBase + r*1024 + col]
// MODE 1: q/k   : per-row L2 norm over the 64-col head chunk, f16 head layout
// MODE 2: v     : scale by Stot^-sigmoid(nc[h]),              f16 head layout
template <int MODE>
__global__ __launch_bounds__(256) void gemm_kernel(
    const _Float16* __restrict__ A, const _Float16* __restrict__ W, int rows,
    int rowsPerBatch, int batchStrideRows, int rowBaseOff,
    float* __restrict__ outF, size_t outBase, _Float16* __restrict__ outH,
    int Stot, int sOff, int sStream, const float* __restrict__ normConst) {
  __shared__ alignas(16) _Float16 As[2][128 * 32];
  __shared__ alignas(16) _Float16 Bs[2][64 * 32];

  const int tid = threadIdx.x;
  const int wave = tid >> 5, lane = tid & 31;
  const int rowBlock = blockIdx.x * 128;
  const int colBlock = blockIdx.y * 64;

  // Pre-resolve the two A-row pointers this thread stages (batch remap).
  const _Float16* aPtr[2];
#pragma unroll
  for (int i = 0; i < 2; ++i) {
    const int slot = tid + i * 256;          // 512 slots: 128 rows x 4 segs
    const int r = rowBlock + (slot >> 2);
    const int rr =
        rowBaseOff + (r / rowsPerBatch) * batchStrideRows + (r % rowsPerBatch);
    aPtr[i] = A + (size_t)rr * 1024 + (slot & 3) * 8;
  }
  const _Float16* wPtr =
      W + (size_t)(colBlock + (tid >> 2)) * 1024 + (tid & 3) * 8;
  const int aOff = (tid >> 2) * 32 + (tid & 3) * 8;  // slot 1 adds 64 rows
  const int bOff = aOff;

  auto issue = [&](int buf, int kk) {
    async_copy16(aPtr[0] + kk, &As[buf][aOff]);
    async_copy16(aPtr[1] + kk, &As[buf][aOff + 64 * 32]);
    async_copy16(wPtr + kk, &Bs[buf][bOff]);
  };

  v8f acc[4] = {};
  issue(0, 0);
  int buf = 0;
  for (int kk = 0; kk < 1024; kk += 32, buf ^= 1) {
    if (kk + 32 < 1024) {
      issue(buf ^ 1, kk + 32);  // overlap next tile with this tile's WMMAs
      wait_async<3>();          // previous 3 async ops (this tile) landed
    } else {
      wait_async<0>();
    }
    __syncthreads();
    const v16h a = load_frag_a(&As[buf][0], wave * 16, lane);
#pragma unroll
    for (int t = 0; t < 4; ++t) {
      const v16h b = load_frag_b(&Bs[buf][0], t * 16, lane);
      acc[t] = WMMA_F16(a, b, acc[t]);
    }
    __syncthreads();
  }

  // ---- epilogue ----
  const int rl = (lane & 16) ? 8 : 0;  // C layout: VGPR e -> M = e (+8 hi half)
  const int cl = lane & 15;

  if (MODE == 0) {
#pragma unroll
    for (int e = 0; e < 8; ++e) {
      const int r = rowBlock + wave * 16 + rl + e;
      float* o = outF + outBase + (size_t)r * 1024 + colBlock + cl;
#pragma unroll
      for (int t = 0; t < 4; ++t) o[t * 16] = acc[t][e];
    }
  } else {
    const int h = blockIdx.y;  // BN==64 -> one head per column block
    float mult[8];
    if (MODE == 1) {
#pragma unroll
      for (int e = 0; e < 8; ++e) {
        float s = 0.f;
#pragma unroll
        for (int t = 0; t < 4; ++t) s += acc[t][e] * acc[t][e];
        s += __shfl_xor(s, 1);   // reduce across the 16-lane C group
        s += __shfl_xor(s, 2);
        s += __shfl_xor(s, 4);
        s += __shfl_xor(s, 8);
        mult[e] = 1.0f / fmaxf(sqrtf(s), 1e-12f);
      }
    } else {
      const float nc = normConst[h];
      const float sg = 1.0f / (1.0f + __expf(-nc));
      const float scale = __powf((float)Stot, -sg);
#pragma unroll
      for (int e = 0; e < 8; ++e) mult[e] = scale;
    }
#pragma unroll
    for (int e = 0; e < 8; ++e) {
      const int r = rowBlock + wave * 16 + rl + e;  // token index in stream
      const int bb = r / sStream, ss = r % sStream;
      const size_t base =
          (((size_t)bb * 16 + h) * (size_t)Stot + (ss + sOff)) * 64 + cl;
#pragma unroll
      for (int t = 0; t < 4; ++t)
        outH[base + t * 16] = (_Float16)(acc[t][e] * mult[e]);
    }
  }
}

// ---------------------------------------------------------------------------
// kv[b,h] = k^T @ v  (64x64), K-dim = Stot tokens.  One block per (b,h).
// Both k and v tiles are transposed into LDS as [dim][seq] so they serve
// directly as A ([M][K]) and B ([N][K]) fragments.  Stores kv^T (f16).
__global__ __launch_bounds__(256) void kv_kernel(
    const _Float16* __restrict__ kbuf, const _Float16* __restrict__ vbuf,
    _Float16* __restrict__ kvT, int Stot) {
  __shared__ alignas(16) _Float16 Ks[64 * 32];
  __shared__ alignas(16) _Float16 Vs[64 * 32];
  const int bh = blockIdx.x;
  const _Float16* kp = kbuf + (size_t)bh * Stot * 64;
  const _Float16* vp = vbuf + (size_t)bh * Stot * 64;
  const int tid = threadIdx.x, wave = tid >> 5, lane = tid & 31;
  const int rb = (wave >> 1) * 16;        // A row block (d0)
  const int cb0 = (wave & 1) * 32;        // B col base (d1): two 16-wide tiles
  const int srow = tid >> 3;              // seq-in-tile 0..31
  const int dseg = (tid & 7) * 8;         // 8 dims per thread

  v8f acc[2] = {};
  for (int kk = 0; kk < Stot; kk += 32) {
    union { uint4 u; _Float16 hh[8]; } tk, tv;
    tk.u = *(const uint4*)(kp + (size_t)(kk + srow) * 64 + dseg);
    tv.u = *(const uint4*)(vp + (size_t)(kk + srow) * 64 + dseg);
#pragma unroll
    for (int j = 0; j < 8; ++j) {
      Ks[(dseg + j) * 32 + srow] = tk.hh[j];
      Vs[(dseg + j) * 32 + srow] = tv.hh[j];
    }
    if (kk + 32 < Stot) {
      __builtin_prefetch(kp + (size_t)(kk + 32 + srow) * 64 + dseg, 0, 1);
      __builtin_prefetch(vp + (size_t)(kk + 32 + srow) * 64 + dseg, 0, 1);
    }
    __syncthreads();
    const v16h a = load_frag_a(Ks, rb, lane);
#pragma unroll
    for (int t = 0; t < 2; ++t) {
      const v16h b = load_frag_b(Vs, cb0 + t * 16, lane);
      acc[t] = WMMA_F16(a, b, acc[t]);
    }
    __syncthreads();
  }
  const int rl = (lane & 16) ? 8 : 0, cl = lane & 15;
  _Float16* o = kvT + (size_t)bh * 4096;
#pragma unroll
  for (int t = 0; t < 2; ++t)
#pragma unroll
    for (int e = 0; e < 8; ++e) {
      const int r = rb + rl + e;
      const int c0 = cb0 + t * 16 + cl;
      o[c0 * 64 + r] = (_Float16)acc[t][e];   // transposed store
    }
}

// ---------------------------------------------------------------------------
// attn[b,h] = q[b,h] (Stot x 64) @ kv[b,h] (64x64); writes merged-head f16
// buffer attnbuf[b][token][h*64+d].  Tiles filled via async global->LDS.
__global__ __launch_bounds__(256) void attn_kernel(
    const _Float16* __restrict__ qbuf, const _Float16* __restrict__ kvT,
    _Float16* __restrict__ attnbuf, int Stot) {
  __shared__ alignas(16) _Float16 Qs[128 * 32];
  __shared__ alignas(16) _Float16 Bs[64 * 32];
  const int bh = blockIdx.y, b = bh >> 4, h = bh & 15;
  const int rowBlock = blockIdx.x * 128;
  const _Float16* qp = qbuf + (size_t)bh * Stot * 64;
  const _Float16* kvp = kvT + (size_t)bh * 4096;
  const int tid = threadIdx.x, wave = tid >> 5, lane = tid & 31;

  v8f acc[4] = {};
  for (int kk = 0; kk < 64; kk += 32) {
#pragma unroll
    for (int i = 0; i < 2; ++i) {
      const int slot = tid + i * 256;
      const int r = rowBlock + (slot >> 2);
      async_copy16(qp + (size_t)r * 64 + kk + (slot & 3) * 8,
                   &Qs[(slot >> 2) * 32 + (slot & 3) * 8]);
    }
    // kvT row n == column n of kv -> Bs[n][k] contiguous copies
    async_copy16(kvp + (tid >> 2) * 64 + kk + (tid & 3) * 8,
                 &Bs[(tid >> 2) * 32 + (tid & 3) * 8]);
    wait_async<0>();
    __syncthreads();
    const v16h a = load_frag_a(Qs, wave * 16, lane);
#pragma unroll
    for (int t = 0; t < 4; ++t) {
      const v16h bfrag = load_frag_b(Bs, t * 16, lane);
      acc[t] = WMMA_F16(a, bfrag, acc[t]);
    }
    __syncthreads();
  }
  const int rl = (lane & 16) ? 8 : 0, cl = lane & 15;
#pragma unroll
  for (int e = 0; e < 8; ++e) {
    const int sg = rowBlock + wave * 16 + rl + e;
    _Float16* o = attnbuf + ((size_t)b * Stot + sg) * 1024 + h * 64 + cl;
#pragma unroll
    for (int t = 0; t < 4; ++t) o[t * 16] = (_Float16)acc[t][e];
  }
}

// ---------------------------------------------------------------------------
extern "C" void kernel_launch(void* const* d_in, const int* in_sizes, int n_in,
                              void* d_out, int out_size, void* d_ws,
                              size_t ws_size, hipStream_t stream) {
  constexpr int D = 1024, H = 16, B = 4, N = 4096, M = 256;
  constexpr int ST = N + M;  // 4352

  const float* x = (const float*)d_in[0];
  const float* c = (const float*)d_in[1];
  const float* norm_const = (const float*)d_in[10];
  float* out = (float*)d_out;

  // workspace carve-out
  size_t off = 0;
  auto alloc = [&](size_t bytes) -> void* {
    void* p = (char*)d_ws + off;
    off += (bytes + 255) & ~(size_t)255;
    return p;
  };
  _Float16* xh = (_Float16*)alloc((size_t)B * N * D * 2);
  _Float16* ch = (_Float16*)alloc((size_t)B * M * D * 2);
  _Float16* wh[8];
  for (int i = 0; i < 8; ++i) wh[i] = (_Float16*)alloc((size_t)D * D * 2);
  _Float16* qb = (_Float16*)alloc((size_t)B * H * ST * 64 * 2);
  _Float16* kb = (_Float16*)alloc((size_t)B * H * ST * 64 * 2);
  _Float16* vb = (_Float16*)alloc((size_t)B * H * ST * 64 * 2);
  _Float16* kvT = (_Float16*)alloc((size_t)B * H * 64 * 64 * 2);
  _Float16* ab = (_Float16*)alloc((size_t)B * ST * D * 2);
  (void)ws_size; (void)in_sizes; (void)n_in; (void)out_size;

  // 1) fp32 -> fp16
  f32_to_f16<<<2048, 256, 0, stream>>>(x, xh, (size_t)B * N * D);
  f32_to_f16<<<256, 256, 0, stream>>>(c, ch, (size_t)B * M * D);
  for (int i = 0; i < 8; ++i)
    f32_to_f16<<<512, 256, 0, stream>>>((const float*)d_in[2 + i], wh[i],
                                        (size_t)D * D);

  const dim3 blk(256);
  const int rowsX = B * N;  // 16384
  const int rowsC = B * M;  // 1024

  // 2) projections (x stream)
  gemm_kernel<1><<<dim3(rowsX / 128, H), blk, 0, stream>>>(
      xh, wh[0], rowsX, rowsX, 0, 0, nullptr, 0, qb, ST, 0, N, norm_const);
  gemm_kernel<1><<<dim3(rowsX / 128, H), blk, 0, stream>>>(
      xh, wh[1], rowsX, rowsX, 0, 0, nullptr, 0, kb, ST, 0, N, norm_const);
  gemm_kernel<2><<<dim3(rowsX / 128, H), blk, 0, stream>>>(
      xh, wh[2], rowsX, rowsX, 0, 0, nullptr, 0, vb, ST, 0, N, norm_const);
  //    projections (c stream)
  gemm_kernel<1><<<dim3(rowsC / 128, H), blk, 0, stream>>>(
      ch, wh[3], rowsC, rowsC, 0, 0, nullptr, 0, qb, ST, N, M, norm_const);
  gemm_kernel<1><<<dim3(rowsC / 128, H), blk, 0, stream>>>(
      ch, wh[4], rowsC, rowsC, 0, 0, nullptr, 0, kb, ST, N, M, norm_const);
  gemm_kernel<2><<<dim3(rowsC / 128, H), blk, 0, stream>>>(
      ch, wh[5], rowsC, rowsC, 0, 0, nullptr, 0, vb, ST, N, M, norm_const);

  // 3) kv = k^T v per (b,h)
  kv_kernel<<<dim3(B * H), blk, 0, stream>>>(kb, vb, kvT, ST);

  // 4) attn = q @ kv
  attn_kernel<<<dim3(ST / 128, B * H), blk, 0, stream>>>(qb, kvT, ab, ST);

  // 5) output projections straight into d_out
  //    x rows live at token offsets [0,4096) of each 4352-row batch
  gemm_kernel<0><<<dim3(rowsX / 128, H), blk, 0, stream>>>(
      ab, wh[6], rowsX, N, ST, 0, out, 0, nullptr, 0, 0, 0, nullptr);
  //    c rows live at token offsets [4096,4352)
  gemm_kernel<0><<<dim3(rowsC / 128, H), blk, 0, stream>>>(
      ab, wh[7], rowsC, M, ST, N, out, (size_t)rowsX * D, nullptr, 0, 0, 0,
      nullptr);
}
